// BiMambaLayer_58566174048583
// MI455X (gfx1250) — matrix operand order repaired and checked by reading, not compiled, not run on hardware
//
#include <hip/hip_runtime.h>
#include <math.h>

#define BATCH   8
#define LSEQ    4096
#define MTOT    (BATCH*LSEQ)    // 32768 token rows
#define DMODEL  512
#define DINNER  1024
#define DSTATE  16
#define DTRANK  32
#define DFF     2048

typedef __attribute__((ext_vector_type(16))) __bf16 v16bf;
typedef __attribute__((ext_vector_type(8)))  float  v8f;
typedef unsigned int u32x4 __attribute__((ext_vector_type(4)));
typedef int          i32x8 __attribute__((ext_vector_type(8)));
typedef int          i32x4 __attribute__((ext_vector_type(4)));

#if __has_builtin(__builtin_amdgcn_tensor_load_to_lds) && \
    __has_builtin(__builtin_amdgcn_s_wait_tensorcnt)
#define USE_TDM 1
#else
#define USE_TDM 0
#endif

__device__ __forceinline__ float bf2f(unsigned short b) {
    unsigned int u = ((unsigned int)b) << 16;
    float f; __builtin_memcpy(&f, &u, 4); return f;
}
__device__ __forceinline__ unsigned short f2bf(float f) {
    unsigned int u; __builtin_memcpy(&u, &f, 4);
    u = (u + 0x7FFFu + ((u >> 16) & 1u)) >> 16;
    return (unsigned short)u;
}

// ---------------------------------------------------------------- converters
__global__ __launch_bounds__(256) void cvt_f32_to_bf16(
    const float* __restrict__ in, unsigned short* __restrict__ out, int n) {
    int i = blockIdx.x * 256 + threadIdx.x;
    if (i < n) out[i] = f2bf(in[i]);
}

#if USE_TDM
// Issue one TDM descriptor moving a [rows x tile_k] bf16 tile (row stride =
// rowlen elements) from global memory into LDS at byte offset lds_off.
// D# fields per ISA 8.3/8.4: group0 = count|lds_addr|global_addr|type=2,
// group1 = data_size(2B) | tensor dims | tile dims | dim0 stride.
__device__ __forceinline__ void tdm_load_tile(
    unsigned int lds_off, const unsigned short* gptr,
    unsigned int rows, unsigned int rows_total,
    unsigned int rowlen, unsigned int tile_k)
{
    unsigned long long ga = (unsigned long long)gptr;
    u32x4 g0;
    g0[0] = 1u;                                   // count=1, user mode
    g0[1] = lds_off;                              // lds_addr (bytes)
    g0[2] = (unsigned int)ga;                     // global_addr[31:0]
    g0[3] = (unsigned int)((ga >> 32) & 0x1FFFFFFu) | (2u << 30); // [56:32]|type=2
    i32x8 g1;
    g1[0] = 0x10000;                              // wg_mask=0, data_size=1 (2B)
    g1[1] = (int)((rowlen & 0xFFFFu) << 16);      // tensor_dim0[15:0] @ bits63:48
    g1[2] = (int)((rowlen >> 16) | ((rows_total & 0xFFFFu) << 16)); // dim0 hi|dim1 lo
    g1[3] = (int)((rows_total >> 16) | (tile_k << 16));            // dim1 hi|tile_dim0
    g1[4] = (int)rows;                            // tile_dim1=rows, tile_dim2=0
    g1[5] = (int)rowlen;                          // tensor_dim0_stride[31:0]
    g1[6] = 0;                                    // stride0 hi | stride1 lo
    g1[7] = 0;                                    // stride1 hi
    i32x4 z4 = {0, 0, 0, 0};
#if __clang_major__ >= 23
    i32x8 z8 = {0, 0, 0, 0, 0, 0, 0, 0};
    __builtin_amdgcn_tensor_load_to_lds(g0, g1, z4, z4, z8, 0);
#else
    __builtin_amdgcn_tensor_load_to_lds(g0, g1, z4, z4, 0);
#endif
}
#endif

// ---------------------------------------------------------------- WMMA GEMM
// C[M,N] = A[M,K](bf16,row-major) * W[N,K](bf16,row-major)^T  (f32 accum)
// Block tile 128x64, K-step 32, 8 waves; each wave does a 32x32 patch as
// 2x2 v_wmma_f32_16x16x32_bf16. Tiles are staged into LDS by the Tensor Data
// Mover (one descriptor per tile, issued by wave 0, gated by s_wait_tensorcnt)
// with a cooperative-load fallback when the TDM builtin is unavailable.
// Epilogue: optional bias, tanh-GELU, f32 and/or bf16 stores.
union FragU { unsigned int u[8]; v16bf v; };

__global__ __launch_bounds__(256) void gemm_bf16_wmma(
    const unsigned short* __restrict__ A,
    const unsigned short* __restrict__ W,
    float* __restrict__ Cf,
    unsigned short* __restrict__ Cb,
    const float* __restrict__ bias,
    int M, int N, int K, int act)
{
    extern __shared__ unsigned short smem[];     // [0,8K): A tile, [8K,12K): B tile
    unsigned short* sA = smem;
    unsigned short* sB = smem + 128 * 32;
    const int tid  = threadIdx.x;
    const int wv   = tid >> 5, lane = tid & 31;
    const int wm   = wv & 3,   wn   = wv >> 2;
    const int kg   = lane >> 4, l16 = lane & 15;
    const int mBase = blockIdx.y * 128;
    const int nBase = blockIdx.x * 64;

    v8f acc[2][2] = {};

    for (int kk = 0; kk < K; kk += 32) {
#if USE_TDM
        if (wv == 0) {
            tdm_load_tile(0u,    A + (size_t)mBase * K + kk, 128u, (unsigned)M,
                          (unsigned)K, 32u);
            tdm_load_tile(8192u, W + (size_t)nBase * K + kk, 64u,  (unsigned)N,
                          (unsigned)K, 32u);
            __builtin_amdgcn_s_wait_tensorcnt(0);
        }
        __syncthreads();
#else
#pragma unroll
        for (int i = 0; i < 2; ++i) {
            int chunk = tid + i * 256;
            int row = chunk >> 2;
            int kc  = (chunk & 3) << 3;
            *(uint4*)(sA + row * 32 + kc) =
                *(const uint4*)(A + (size_t)(mBase + row) * K + kk + kc);
        }
        {
            int row = tid >> 2;
            int kc  = (tid & 3) << 3;
            *(uint4*)(sB + row * 32 + kc) =
                *(const uint4*)(W + (size_t)(nBase + row) * K + kk + kc);
        }
        __syncthreads();
#endif

        FragU a[2], b[2];
        // A 16x32 bf16 layout: lane-half gives K-group of 8; V0-3 hold K 0-7(+8),
        // V4-7 hold K 16-23(+8) (ISA 7.12.2).
#pragma unroll
        for (int mt = 0; mt < 2; ++mt) {
            int row = wm * 32 + mt * 16 + l16;
#pragma unroll
            for (int p = 0; p < 8; ++p) {
                int kb = ((p < 4) ? (p * 2) : (16 + (p - 4) * 2)) + kg * 8;
                a[mt].u[p] = *(const unsigned int*)(sA + row * 32 + kb);
            }
        }
        // B 32x16: same K layout as A (lane-half gives K-group of 8; V0-3 K 0-7(+8), V4-7 K 16-23(+8)), col = lane%16.
#pragma unroll
        for (int nt = 0; nt < 2; ++nt) {
            int col = wn * 32 + nt * 16 + l16;
#pragma unroll
            for (int p = 0; p < 8; ++p) {
                int kb = ((p < 4) ? (p * 2) : (16 + (p - 4) * 2)) + kg * 8;
                b[nt].u[p] = *(const unsigned int*)(sB + col * 32 + kb);
            }
        }
#pragma unroll
        for (int mt = 0; mt < 2; ++mt)
#pragma unroll
            for (int nt = 0; nt < 2; ++nt)
                acc[mt][nt] = __builtin_amdgcn_wmma_f32_16x16x32_bf16(
                    false, a[mt].v, false, b[nt].v, (short)0, acc[mt][nt],
                    false, false);
        __syncthreads();
    }

    // epilogue: C/D layout — lane half selects M offset 0/8, VGPR r = M row
#pragma unroll
    for (int mt = 0; mt < 2; ++mt) {
#pragma unroll
        for (int nt = 0; nt < 2; ++nt) {
            int gn = nBase + wn * 32 + nt * 16 + l16;
            float bv = bias ? bias[gn] : 0.f;
#pragma unroll
            for (int r = 0; r < 8; ++r) {
                int gm = mBase + wm * 32 + mt * 16 + r + kg * 8;
                float v = acc[mt][nt][r] + bv;
                if (act == 1) {  // tanh-GELU (jax.nn.gelu approximate)
                    float t = 0.7978845608f * (v + 0.044715f * v * v * v);
                    v = 0.5f * v * (1.f + tanhf(t));
                }
                size_t idx = (size_t)gm * N + gn;
                if (Cf) Cf[idx] = v;
                if (Cb) Cb[idx] = f2bf(v);
            }
        }
    }
}

// ------------------------------------------------- depthwise conv(4) + SiLU
// dir=0: causal taps (u-3..u); dir=1: anti-causal (u..u+3) — implements the
// flipped-sequence conv at original indices.
__global__ __launch_bounds__(256) void conv_silu_kernel(
    const float* __restrict__ xz, const float* __restrict__ cw,
    const float* __restrict__ cb, unsigned short* __restrict__ xc_bf, int dir)
{
    long long li = (long long)blockIdx.x * 256 + threadIdx.x;
    int d = (int)(li & (DINNER - 1));
    int u = (int)((li >> 10) & (LSEQ - 1));
    int b = (int)(li >> 22);
    float acc = cb[d];
#pragma unroll
    for (int j = 0; j < 4; ++j) {
        int off = dir ? (3 - j) : (j - 3);
        int uu = u + off;
        if (uu >= 0 && uu < LSEQ)
            acc += cw[d * 4 + j] * xz[(size_t)(b * LSEQ + uu) * 2048 + d];
    }
    float s = acc / (1.f + __expf(-acc));
    xc_bf[(size_t)(b * LSEQ + u) * DINNER + d] = f2bf(s);
}

// ------------------------------------------------------------ selective scan
// One lane per (b,d); h[16], A row, dt_w row live in VGPRs; per-step dbc row
// (dt(32)|B(16)|C(16)) staged in LDS. Fuses the dt projection + softplus,
// the D skip, and the SiLU(z) gate; emits bf16 y for the out-projection GEMM.
__global__ __launch_bounds__(64) void scan_kernel(
    const float* __restrict__ dbc, const unsigned short* __restrict__ xc_bf,
    const float* __restrict__ xz, const float* __restrict__ dt_w,
    const float* __restrict__ dt_b, const float* __restrict__ A_log,
    const float* __restrict__ Dp, unsigned short* __restrict__ y_bf, int dir)
{
    int b = blockIdx.x >> 4;
    int d = (blockIdx.x & 15) * 64 + threadIdx.x;

    float dtw[32];
#pragma unroll
    for (int r = 0; r < 32; ++r) dtw[r] = dt_w[d * 32 + r];
    float Ar[16], h[16];
#pragma unroll
    for (int n = 0; n < 16; ++n) { Ar[n] = -__expf(A_log[d * 16 + n]); h[n] = 0.f; }
    float dtb = dt_b[d], Dd = Dp[d];

    __shared__ float s[64];
    for (int st = 0; st < LSEQ; ++st) {
        int u = dir ? (LSEQ - 1 - st) : st;
        size_t row = (size_t)b * LSEQ + u;
        s[threadIdx.x] = dbc[row * 64 + threadIdx.x];
        __syncthreads();
        float dtv = dtb;
#pragma unroll
        for (int r = 0; r < 32; ++r) dtv += s[r] * dtw[r];
        dtv = (dtv > 20.f) ? dtv : __logf(1.f + __expf(dtv));  // softplus
        float xv = bf2f(xc_bf[row * DINNER + d]);
        float zv = xz[row * 2048 + DINNER + d];
        float dtx = dtv * xv;
        float y = 0.f;
#pragma unroll
        for (int n = 0; n < 16; ++n) {
            h[n] = __expf(dtv * Ar[n]) * h[n] + dtx * s[32 + n];
            y += h[n] * s[48 + n];
        }
        y = (y + xv * Dd) * (zv / (1.f + __expf(-zv)));
        y_bf[row * DINNER + d] = f2bf(y);
        __syncthreads();
    }
}

// ------------------------------------------------------------ layernorm ops
__device__ __forceinline__ float wave_sum(float v) {
#pragma unroll
    for (int m = 16; m >= 1; m >>= 1) v += __shfl_xor(v, m, 32);
    return v;
}

// h = 0.5*(LN(x+mf; gf,bf) + LN(x+mb; gb,bb)); writes f32 + bf16
__global__ __launch_bounds__(256) void ln_combine_kernel(
    const float* __restrict__ x, const float* __restrict__ mf,
    const float* __restrict__ mb,
    const float* __restrict__ gf, const float* __restrict__ bf_,
    const float* __restrict__ gb, const float* __restrict__ bb,
    float* __restrict__ h, unsigned short* __restrict__ hbf)
{
    int row  = blockIdx.x * 8 + (threadIdx.x >> 5);
    int lane = threadIdx.x & 31;
    float vf[16], vb[16], sf = 0.f, sb = 0.f;
#pragma unroll
    for (int i = 0; i < 16; ++i) {
        size_t idx = (size_t)row * DMODEL + lane + i * 32;
        float xv = x[idx];
        vf[i] = xv + mf[idx];
        vb[i] = xv + mb[idx];
        sf += vf[i]; sb += vb[i];
    }
    sf = wave_sum(sf) * (1.f / DMODEL);
    sb = wave_sum(sb) * (1.f / DMODEL);
    float qf = 0.f, qb = 0.f;
#pragma unroll
    for (int i = 0; i < 16; ++i) {
        float a = vf[i] - sf, c = vb[i] - sb;
        qf += a * a; qb += c * c;
    }
    qf = wave_sum(qf) * (1.f / DMODEL);
    qb = wave_sum(qb) * (1.f / DMODEL);
    float rf = rsqrtf(qf + 1e-5f), rb = rsqrtf(qb + 1e-5f);
#pragma unroll
    for (int i = 0; i < 16; ++i) {
        int c = lane + i * 32;
        size_t idx = (size_t)row * DMODEL + c;
        float hv = 0.5f * (((vf[i] - sf) * rf * gf[c] + bf_[c]) +
                           ((vb[i] - sb) * rb * gb[c] + bb[c]));
        h[idx] = hv;
        hbf[idx] = f2bf(hv);
    }
}

// out = LN(h + ff; g,b)
__global__ __launch_bounds__(256) void ln_final_kernel(
    const float* __restrict__ h, const float* __restrict__ ff,
    const float* __restrict__ g, const float* __restrict__ bta,
    float* __restrict__ out)
{
    int row  = blockIdx.x * 8 + (threadIdx.x >> 5);
    int lane = threadIdx.x & 31;
    float v[16], s = 0.f;
#pragma unroll
    for (int i = 0; i < 16; ++i) {
        size_t idx = (size_t)row * DMODEL + lane + i * 32;
        v[i] = h[idx] + ff[idx];
        s += v[i];
    }
    s = wave_sum(s) * (1.f / DMODEL);
    float q = 0.f;
#pragma unroll
    for (int i = 0; i < 16; ++i) { float a = v[i] - s; q += a * a; }
    q = wave_sum(q) * (1.f / DMODEL);
    float r = rsqrtf(q + 1e-5f);
#pragma unroll
    for (int i = 0; i < 16; ++i) {
        int c = lane + i * 32;
        size_t idx = (size_t)row * DMODEL + c;
        out[idx] = (v[i] - s) * r * g[c] + bta[c];
    }
}

// ---------------------------------------------------------------- launcher
extern "C" void kernel_launch(void* const* d_in, const int* in_sizes, int n_in,
                              void* d_out, int out_size, void* d_ws, size_t ws_size,
                              hipStream_t stream)
{
    (void)in_sizes; (void)n_in; (void)out_size; (void)ws_size;
    const float* x        = (const float*)d_in[0];
    const float* in_w[2]   = {(const float*)d_in[1],  (const float*)d_in[10]};
    const float* conv_w[2] = {(const float*)d_in[2],  (const float*)d_in[11]};
    const float* conv_b[2] = {(const float*)d_in[3],  (const float*)d_in[12]};
    const float* xproj_w[2]= {(const float*)d_in[4],  (const float*)d_in[13]};
    const float* dt_w[2]   = {(const float*)d_in[5],  (const float*)d_in[14]};
    const float* dt_b[2]   = {(const float*)d_in[6],  (const float*)d_in[15]};
    const float* A_log[2]  = {(const float*)d_in[7],  (const float*)d_in[16]};
    const float* Dp[2]     = {(const float*)d_in[8],  (const float*)d_in[17]};
    const float* out_w[2]  = {(const float*)d_in[9],  (const float*)d_in[18]};
    const float* ln_f_g  = (const float*)d_in[19];
    const float* ln_f_b  = (const float*)d_in[20];
    const float* ln_b_g  = (const float*)d_in[21];
    const float* ln_b_b  = (const float*)d_in[22];
    const float* ln_ff_g = (const float*)d_in[23];
    const float* ln_ff_b = (const float*)d_in[24];
    const float* ffn_w1  = (const float*)d_in[25];
    const float* ffn_b1  = (const float*)d_in[26];
    const float* ffn_w2  = (const float*)d_in[27];
    const float* ffn_b2  = (const float*)d_in[28];

    char* ws = (char*)d_ws;
    size_t off = 0;
    auto alloc = [&](size_t bytes) -> char* {
        char* p = ws + off;
        off = (off + bytes + 255) & ~(size_t)255;
        return p;
    };

    unsigned short* xbf   = (unsigned short*)alloc((size_t)MTOT * DMODEL * 2);
    unsigned short* w_in[2], *w_out[2], *w_xp[2];
    for (int i = 0; i < 2; ++i) {
        w_in[i]  = (unsigned short*)alloc((size_t)(2 * DINNER) * DMODEL * 2);
        w_out[i] = (unsigned short*)alloc((size_t)DMODEL * DINNER * 2);
        w_xp[i]  = (unsigned short*)alloc((size_t)64 * DINNER * 2);
    }
    unsigned short* w_f1 = (unsigned short*)alloc((size_t)DFF * DMODEL * 2);
    unsigned short* w_f2 = (unsigned short*)alloc((size_t)DMODEL * DFF * 2);
    float*          xz   = (float*)alloc((size_t)MTOT * 2048 * 4);            // 256 MB
    unsigned short* xcbf = (unsigned short*)alloc((size_t)MTOT * DINNER * 2); // 64 MB
    float*          dbc  = (float*)alloc((size_t)MTOT * 64 * 4);              // 8 MB
    unsigned short* ybf  = (unsigned short*)alloc((size_t)MTOT * DINNER * 2); // 64 MB
    float* mres[2];
    mres[0] = (float*)alloc((size_t)MTOT * DMODEL * 4);
    mres[1] = (float*)alloc((size_t)MTOT * DMODEL * 4);
    // aliases (lifetimes disjoint from their hosts)
    unsigned short* gbf  = (unsigned short*)xz;                               // 128 MB in xz
    float* ff2 = (float*)((char*)xz + (((size_t)MTOT * DFF * 2 + 255) & ~(size_t)255));
    float*          hbuf = (float*)xcbf;                                      // 64 MB
    unsigned short* hbf  = ybf;                                               // 32 MB

    auto cvt = [&](const float* src, unsigned short* dst, size_t n) {
        cvt_f32_to_bf16<<<dim3((unsigned)((n + 255) / 256)), dim3(256), 0, stream>>>(
            src, dst, (int)n);
    };
    cvt(x, xbf, (size_t)MTOT * DMODEL);
    for (int i = 0; i < 2; ++i) {
        cvt(in_w[i],    w_in[i],  (size_t)(2 * DINNER) * DMODEL);
        cvt(out_w[i],   w_out[i], (size_t)DMODEL * DINNER);
        cvt(xproj_w[i], w_xp[i],  (size_t)64 * DINNER);
    }
    cvt(ffn_w1, w_f1, (size_t)DFF * DMODEL);
    cvt(ffn_w2, w_f2, (size_t)DMODEL * DFF);

    const unsigned gemm_lds = (128 * 32 + 64 * 32) * 2;  // 12 KB

    for (int dir = 0; dir < 2; ++dir) {
        // xz = x @ in_w^T   [32768 x 2048]
        gemm_bf16_wmma<<<dim3(2048 / 64, MTOT / 128), 256, gemm_lds, stream>>>(
            xbf, w_in[dir], xz, nullptr, nullptr, MTOT, 2048, DMODEL, 0);
        // depthwise conv + SiLU -> xc (bf16)
        conv_silu_kernel<<<dim3((MTOT * DINNER) / 256), 256, 0, stream>>>(
            xz, conv_w[dir], conv_b[dir], xcbf, dir);
        // dbc = xc @ xproj^T   [32768 x 64]
        gemm_bf16_wmma<<<dim3(1, MTOT / 128), 256, gemm_lds, stream>>>(
            xcbf, w_xp[dir], dbc, nullptr, nullptr, MTOT, 64, DINNER, 0);
        // fused dt-proj + softplus + SSM scan + D skip + SiLU(z) gate
        scan_kernel<<<dim3(BATCH * 16), 64, 0, stream>>>(
            dbc, xcbf, xz, dt_w[dir], dt_b[dir], A_log[dir], Dp[dir], ybf, dir);
        // mamba out = y @ out_w^T   [32768 x 512]
        gemm_bf16_wmma<<<dim3(DMODEL / 64, MTOT / 128), 256, gemm_lds, stream>>>(
            ybf, w_out[dir], mres[dir], nullptr, nullptr, MTOT, DMODEL, DINNER, 0);
    }

    // h = 0.5*(LN(x+mf) + LN(x+mb))
    ln_combine_kernel<<<dim3(MTOT / 8), 256, 0, stream>>>(
        x, mres[0], mres[1], ln_f_g, ln_f_b, ln_b_g, ln_b_b, hbuf, hbf);
    // FFN: gelu(h@w1^T + b1) @ w2^T + b2
    gemm_bf16_wmma<<<dim3(DFF / 64, MTOT / 128), 256, gemm_lds, stream>>>(
        hbf, w_f1, nullptr, gbf, ffn_b1, MTOT, DFF, DMODEL, 1);
    gemm_bf16_wmma<<<dim3(DMODEL / 64, MTOT / 128), 256, gemm_lds, stream>>>(
        gbf, w_f2, ff2, nullptr, ffn_b2, MTOT, DMODEL, DFF, 0);
    // out = LN(h + ff)
    ln_final_kernel<<<dim3(MTOT / 8), 256, 0, stream>>>(
        hbuf, ff2, ln_ff_g, ln_ff_b, (float*)d_out);
}
